// MambaBlock_34797825032197
// MI455X (gfx1250) — compile-verified, matrix operand order checked by reading, not run on hardware
//
#include <hip/hip_runtime.h>
#include <hip/hip_fp16.h>
#include <hip/hip_bf16.h>

// ---- problem constants (from reference) ----
#define DIMC     128
#define DEPTH    8
#define D_INNER  256
#define D_STATE  16
#define D_CONV   4
#define DT_RANK  8
#define LSEQ     4096            // 64*64
#define BATCH    2
#define NTOK     (BATCH * LSEQ)  // 8192 tokens
#define XP_N     40              // DT_RANK + 2*D_STATE
#define XP_NPAD  64              // padded to one 4-subtile WMMA strip
#define NCHUNK   16              // scan chunks per sequence
#define CHUNK    (LSEQ / NCHUNK) // 256

typedef __attribute__((ext_vector_type(16))) _Float16 v16h;
typedef __attribute__((ext_vector_type(8)))  _Float16 v8h;
typedef __attribute__((ext_vector_type(8)))  float    v8f;

#define SHUF16(lo, hi) __builtin_shufflevector((lo), (hi), \
        0,1,2,3,4,5,6,7,8,9,10,11,12,13,14,15)

// ============================================================
// (B,C,H,W) -> token-major f16 (B*L, C)
// ============================================================
__global__ void transpose_in_kernel(const float* __restrict__ x, _Float16* __restrict__ h16)
{
    int idx = blockIdx.x * blockDim.x + threadIdx.x;      // over NTOK*DIMC
    if (idx >= NTOK * DIMC) return;
    int c = idx & (DIMC - 1);
    int t = idx >> 7;                                     // b*L + l
    int b = t >> 12;
    int l = t & (LSEQ - 1);
    h16[idx] = (_Float16)x[((size_t)(b * DIMC + c)) * LSEQ + l];
}

// ============================================================
// Pack (DEPTH, Nreal, K) f32 weights -> (DEPTH, Npad, K) f16, zero padded rows.
// ============================================================
__global__ void pack_w_kernel(const float* __restrict__ src, _Float16* __restrict__ dst,
                              int Nreal, int Npad, int K, int total)
{
    int idx = blockIdx.x * blockDim.x + threadIdx.x;      // over DEPTH*Npad*K
    if (idx >= total) return;
    int k    = idx % K;
    int rest = idx / K;
    int n    = rest % Npad;
    int lyr  = rest / Npad;
    float v = (n < Nreal) ? src[((size_t)lyr * Nreal + n) * K + k] : 0.0f;
    dst[idx] = (_Float16)v;
}

// ============================================================
// WMMA GEMM, all-f16 operands:  C[M,N] = A[M,K] * Bw[N,K]^T
// One wave per 32x64 strip: 2 M-tiles x 4 N-tiles = 8 WMMAs per k-step.
// K/LDA/LDB/LDC are compile-time so all 12 loads per k-step use just TWO
// base address pairs + instruction immediate offsets (row strides 16*LDA,
// j*16*LDB fit the 24-bit IOFFSET) -> no address spills.
// CDNA5 wave32 layouts:
//   A: lane = hf*16 + m,  element e -> K = k0 + 8*hf + e + (e>=8 ? 8 : 0)
//   B: lane = hf*16 + n,  same K mapping
//   D: v8f element r, lane -> (M = r + 8*hf, N = lane&15)
// M % 32 == 0, K % 32 == 0, N % 64 == 0 -> no guards, EXEC all-ones.
// ============================================================
template<int K, int LDA, int LDB, int LDC, bool MIRROR>
__global__ void wmma_gemm_kernel(const _Float16* __restrict__ A,
                                 const _Float16* __restrict__ Bw,
                                 float* __restrict__ C,
                                 _Float16* __restrict__ C16,
                                 int M, int Nstrips)
{
    const int wavesPerBlock = blockDim.x >> 5;
    const int strip = blockIdx.x * wavesPerBlock + (threadIdx.x >> 5);
    const int total = (M >> 5) * Nstrips;
    if (strip >= total) return;                 // wave-uniform

    const int mblk   = strip / Nstrips;         // 32-row block
    const int nstrip = strip - mblk * Nstrips;  // 64-col strip

    const int lane = threadIdx.x & 31;
    const int lm   = lane & 15;
    const int hf   = lane >> 4;

    // two per-lane base addresses; all other operands via immediate offsets
    const _Float16* Aaddr = A  + (size_t)(mblk * 32 + lm) * LDA + hf * 8;
    const _Float16* Baddr = Bw + (size_t)(nstrip * 64 + lm) * LDB + hf * 8;

    constexpr int A1 = (16 * LDA) / 8;          // v8h units to next M-tile row set
    constexpr int B1 = (16 * LDB) / 8;          // v8h units to next N-subtile row set

    v8f acc0[4] = {};   // M-tile 0
    v8f acc1[4] = {};   // M-tile 1

    for (int k0 = 0; k0 < K; k0 += 32) {
        const v8h* pa = reinterpret_cast<const v8h*>(Aaddr);
        const v8h* pb = reinterpret_cast<const v8h*>(Baddr);

        v16h av0 = SHUF16(pa[0],          pa[2]);
        v16h av1 = SHUF16(pa[A1],         pa[A1 + 2]);
        v16h bv0 = SHUF16(pb[0],          pb[2]);
        v16h bv1 = SHUF16(pb[B1],         pb[B1 + 2]);
        v16h bv2 = SHUF16(pb[2 * B1],     pb[2 * B1 + 2]);
        v16h bv3 = SHUF16(pb[3 * B1],     pb[3 * B1 + 2]);

        if (k0 + 32 < K) {                      // prefetch next k-step rows
            __builtin_prefetch(Aaddr + 32, 0, 1);
            __builtin_prefetch(Baddr + 32, 0, 1);
        }

        acc0[0] = __builtin_amdgcn_wmma_f32_16x16x32_f16(false, av0, false, bv0, (short)0, acc0[0], false, false);
        acc1[0] = __builtin_amdgcn_wmma_f32_16x16x32_f16(false, av1, false, bv0, (short)0, acc1[0], false, false);
        acc0[1] = __builtin_amdgcn_wmma_f32_16x16x32_f16(false, av0, false, bv1, (short)0, acc0[1], false, false);
        acc1[1] = __builtin_amdgcn_wmma_f32_16x16x32_f16(false, av1, false, bv1, (short)0, acc1[1], false, false);
        acc0[2] = __builtin_amdgcn_wmma_f32_16x16x32_f16(false, av0, false, bv2, (short)0, acc0[2], false, false);
        acc1[2] = __builtin_amdgcn_wmma_f32_16x16x32_f16(false, av1, false, bv2, (short)0, acc1[2], false, false);
        acc0[3] = __builtin_amdgcn_wmma_f32_16x16x32_f16(false, av0, false, bv3, (short)0, acc0[3], false, false);
        acc1[3] = __builtin_amdgcn_wmma_f32_16x16x32_f16(false, av1, false, bv3, (short)0, acc1[3], false, false);

        Aaddr += 32;
        Baddr += 32;
    }

    const int rowbase = mblk * 32 + 8 * hf;
    const int colbase = nstrip * 64 + lm;
#pragma unroll
    for (int j = 0; j < 4; ++j) {
#pragma unroll
        for (int r = 0; r < 8; ++r) {
            const size_t o0 = (size_t)(rowbase + r)      * LDC + colbase + j * 16;
            const size_t o1 = (size_t)(rowbase + 16 + r) * LDC + colbase + j * 16;
            C[o0] = acc0[j][r];
            C[o1] = acc1[j][r];
            if (MIRROR) {
                C16[o0] = (_Float16)acc0[j][r];
                C16[o1] = (_Float16)acc1[j][r];
            }
        }
    }
}

// ============================================================
// Depthwise causal conv (width 4) + bias + SiLU; emits f32 (scan) + f16 (GEMM A)
// ============================================================
__global__ void conv_silu_kernel(const float* __restrict__ xz,
                                 const float* __restrict__ cw,
                                 const float* __restrict__ cb,
                                 float* __restrict__ xc,
                                 _Float16* __restrict__ xc16)
{
    int idx = blockIdx.x * blockDim.x + threadIdx.x;   // over NTOK*D_INNER
    if (idx >= NTOK * D_INNER) return;
    int d = idx & (D_INNER - 1);
    int t = idx >> 8;
    int b = t >> 12;
    int l = t & (LSEQ - 1);

    float acc = cb[d];
#pragma unroll
    for (int j = 0; j < D_CONV; ++j) {
        int ll = l - (D_CONV - 1) + j;
        if (ll >= 0)
            acc += cw[d * D_CONV + j] *
                   xz[((size_t)(b * LSEQ + ll)) * (2 * D_INNER) + d];
    }
    float v = acc / (1.0f + __expf(-acc));
    xc[(size_t)t * D_INNER + d]   = v;
    xc16[(size_t)t * D_INNER + d] = (_Float16)v;
}

// ============================================================
// delta = softplus(dt @ dtw^T + dtb),  dt = xdbl[:, 0:8]  (ldc = 64)
// ============================================================
__global__ void delta_kernel(const float* __restrict__ xdbl,
                             const float* __restrict__ dtw,
                             const float* __restrict__ dtb,
                             float* __restrict__ delta)
{
    int idx = blockIdx.x * blockDim.x + threadIdx.x;   // over NTOK*D_INNER
    if (idx >= NTOK * D_INNER) return;
    int d = idx & (D_INNER - 1);
    int t = idx >> 8;

    const float* dt = xdbl + (size_t)t * XP_NPAD;
    float acc = dtb[d];
#pragma unroll
    for (int r = 0; r < DT_RANK; ++r)
        acc += dt[r] * dtw[d * DT_RANK + r];

    delta[(size_t)t * D_INNER + d] = (acc > 20.0f) ? acc : log1pf(__expf(acc));
}

// ============================================================
// Chunked selective scan.
// Group g = (b*256+d)*16 + chunk, 16 lanes per group (lane = state n).
// Pass 1: per-chunk (prod dA, local end-state) -> carryA/carryS
// ============================================================
__global__ void scan_local_kernel(const float* __restrict__ xc,
                                  const float* __restrict__ xdbl,
                                  const float* __restrict__ delta,
                                  const float* __restrict__ A_log,
                                  float* __restrict__ carryA,
                                  float* __restrict__ carryS)
{
    const int lane = threadIdx.x & 31;
    const int n    = lane & 15;
    const int g    = (((blockIdx.x * blockDim.x + threadIdx.x) >> 5) << 1) | (lane >> 4);
    if (g >= BATCH * D_INNER * NCHUNK) return;

    const int chunk = g & (NCHUNK - 1);
    const int cd    = g >> 4;                 // b*256 + d
    const int d     = cd & (D_INNER - 1);
    const int b     = cd >> 8;

    const float An = -__expf(A_log[d * D_STATE + n]);
    float s = 0.0f, P = 1.0f;

    const size_t t0 = (size_t)b * LSEQ + chunk * CHUNK;
    for (int l = 0; l < CHUNK; ++l) {
        const size_t t  = t0 + l;
        const float  u  = xc   [t * D_INNER + d];
        const float  dl = delta[t * D_INNER + d];
        const float  Bn = xdbl [t * XP_NPAD + DT_RANK + n];
        const float  a  = __expf(dl * An);
        s = s * a + (dl * u) * Bn;
        P *= a;
    }
    carryA[(size_t)g * D_STATE + n] = P;
    carryS[(size_t)g * D_STATE + n] = s;
}

// Pass 2: sequential carry over the 16 chunks (per (b,d,n) thread)
__global__ void scan_carry_kernel(const float* __restrict__ carryA,
                                  const float* __restrict__ carryS,
                                  float* __restrict__ inState)
{
    int tid = blockIdx.x * blockDim.x + threadIdx.x;   // over B*D_INNER*D_STATE
    if (tid >= BATCH * D_INNER * D_STATE) return;
    int n  = tid & (D_STATE - 1);
    int cd = tid >> 4;

    float h = 0.0f;
#pragma unroll
    for (int chunk = 0; chunk < NCHUNK; ++chunk) {
        size_t o = ((size_t)cd * NCHUNK + chunk) * D_STATE + n;
        inState[o] = h;
        h = h * carryA[o] + carryS[o];
    }
}

// Pass 3: replay each chunk from its incoming state, reduce over n,
// fuse (+u*D)*silu(z), emit y in f16 for the out_proj WMMA.
__global__ void scan_final_kernel(const float* __restrict__ xc,
                                  const float* __restrict__ xdbl,
                                  const float* __restrict__ delta,
                                  const float* __restrict__ xz,
                                  const float* __restrict__ A_log,
                                  const float* __restrict__ D_skip,
                                  const float* __restrict__ inState,
                                  _Float16* __restrict__ y16)
{
    const int lane = threadIdx.x & 31;
    const int n    = lane & 15;
    const int g    = (((blockIdx.x * blockDim.x + threadIdx.x) >> 5) << 1) | (lane >> 4);
    if (g >= BATCH * D_INNER * NCHUNK) return;

    const int chunk = g & (NCHUNK - 1);
    const int cd    = g >> 4;
    const int d     = cd & (D_INNER - 1);
    const int b     = cd >> 8;

    const float An = -__expf(A_log[d * D_STATE + n]);
    const float Dv = D_skip[d];
    float s = inState[(size_t)g * D_STATE + n];

    const size_t t0 = (size_t)b * LSEQ + chunk * CHUNK;
    for (int l = 0; l < CHUNK; ++l) {
        const size_t t  = t0 + l;
        const float  u  = xc   [t * D_INNER + d];
        const float  dl = delta[t * D_INNER + d];
        const float  Bn = xdbl [t * XP_NPAD + DT_RANK + n];
        const float  Cn = xdbl [t * XP_NPAD + DT_RANK + D_STATE + n];

        s = s * __expf(dl * An) + (dl * u) * Bn;

        float part = s * Cn;
        part += __shfl_xor(part, 1);
        part += __shfl_xor(part, 2);
        part += __shfl_xor(part, 4);
        part += __shfl_xor(part, 8);

        if (n == 0) {
            const float z  = xz[t * (2 * D_INNER) + D_INNER + d];
            const float sz = z / (1.0f + __expf(-z));
            y16[t * D_INNER + d] = (_Float16)((part + u * Dv) * sz);
        }
    }
}

// ============================================================
// RMS norm over C=128 per token + transpose back to (B,C,H,W)
// ============================================================
__global__ void rmsnorm_out_kernel(const float* __restrict__ h,
                                   const float* __restrict__ rms_w,
                                   float* __restrict__ out)
{
    const int t = blockIdx.x;          // token
    const int c = threadIdx.x;         // channel, blockDim = 128
    float v = h[(size_t)t * DIMC + c];

    float sq = v * v;
#pragma unroll
    for (int m = 16; m >= 1; m >>= 1) sq += __shfl_xor(sq, m);

    __shared__ float red[4];
    if ((threadIdx.x & 31) == 0) red[threadIdx.x >> 5] = sq;
    __syncthreads();
    const float tot = red[0] + red[1] + red[2] + red[3];
    const float r = rsqrtf(tot / (float)DIMC + 1e-6f);

    const int b = t >> 12;
    const int l = t & (LSEQ - 1);
    out[((size_t)(b * DIMC + c)) * LSEQ + l] = v * r * rms_w[c];
}

// ============================================================
extern "C" void kernel_launch(void* const* d_in, const int* in_sizes, int n_in,
                              void* d_out, int out_size, void* d_ws, size_t ws_size,
                              hipStream_t stream)
{
    const float* x       = (const float*)d_in[0];
    const float* in_w    = (const float*)d_in[1];   // (8, 512, 128)
    const float* conv_w  = (const float*)d_in[2];   // (8, 256, 4)
    const float* conv_b  = (const float*)d_in[3];   // (8, 256)
    const float* xp_w    = (const float*)d_in[4];   // (8, 40, 256)
    const float* dt_w    = (const float*)d_in[5];   // (8, 256, 8)
    const float* dt_b    = (const float*)d_in[6];   // (8, 256)
    const float* A_log   = (const float*)d_in[7];   // (8, 256, 16)
    const float* D_skip  = (const float*)d_in[8];   // (8, 256)
    const float* out_w   = (const float*)d_in[9];   // (8, 128, 256)
    const float* rms_w   = (const float*)d_in[10];  // (128,)
    float* out = (float*)d_out;

    // ---- workspace carve-up (byte offsets, 256B aligned) ----
    char* ws = (char*)d_ws;
    size_t off = 0;
    auto alloc = [&](size_t bytes) {
        char* p = ws + off;
        off += (bytes + 255) & ~(size_t)255;
        return (void*)p;
    };
    _Float16* h16   = (_Float16*)alloc((size_t)NTOK * DIMC * 2);
    float*    hb    = (float*)   alloc((size_t)NTOK * DIMC * 4);
    _Float16* w_in  = (_Float16*)alloc((size_t)DEPTH * 512 * DIMC * 2);
    _Float16* w_xp  = (_Float16*)alloc((size_t)DEPTH * XP_NPAD * D_INNER * 2);
    _Float16* w_out = (_Float16*)alloc((size_t)DEPTH * DIMC * D_INNER * 2);
    float*    xz    = (float*)   alloc((size_t)NTOK * 2 * D_INNER * 4);
    float*    xc    = (float*)   alloc((size_t)NTOK * D_INNER * 4);
    _Float16* xc16  = (_Float16*)alloc((size_t)NTOK * D_INNER * 2);
    float*    xdbl  = (float*)   alloc((size_t)NTOK * XP_NPAD * 4);
    float*    dlt   = (float*)   alloc((size_t)NTOK * D_INNER * 4);
    _Float16* y16   = (_Float16*)alloc((size_t)NTOK * D_INNER * 2);
    float*    carA  = (float*)   alloc((size_t)BATCH * D_INNER * NCHUNK * D_STATE * 4);
    float*    carS  = (float*)   alloc((size_t)BATCH * D_INNER * NCHUNK * D_STATE * 4);
    float*    inSt  = (float*)   alloc((size_t)BATCH * D_INNER * NCHUNK * D_STATE * 4);
    (void)ws_size;

    const int EBLK = 256;
    auto blocks = [](int n, int blk) { return (n + blk - 1) / blk; };

    // ---- one-time (per launch) operand packing ----
    transpose_in_kernel<<<blocks(NTOK * DIMC, EBLK), EBLK, 0, stream>>>(x, h16);
    {
        int t1 = DEPTH * 512 * DIMC;
        pack_w_kernel<<<blocks(t1, EBLK), EBLK, 0, stream>>>(in_w,  w_in,  512, 512, DIMC, t1);
        int t2 = DEPTH * XP_NPAD * D_INNER;
        pack_w_kernel<<<blocks(t2, EBLK), EBLK, 0, stream>>>(xp_w,  w_xp,  XP_N, XP_NPAD, D_INNER, t2);
        int t3 = DEPTH * DIMC * D_INNER;
        pack_w_kernel<<<blocks(t3, EBLK), EBLK, 0, stream>>>(out_w, w_out, DIMC, DIMC, D_INNER, t3);
    }

    const int scanGroups = BATCH * D_INNER * NCHUNK;          // 8192
    const int scanBlocks = (scanGroups / 2 * 32) / EBLK;      // 512

    for (int i = 0; i < DEPTH; ++i) {
        // --- in_proj: (8192,128) x (512,128)^T -> f32 xz (8192,512) ---
        {
            const int strips = (NTOK / 32) * (512 / 64);      // 2048
            wmma_gemm_kernel<DIMC, DIMC, DIMC, 512, false>
                <<<blocks(strips, 4), 128, 0, stream>>>(
                h16, w_in + (size_t)i * 512 * DIMC, xz, nullptr, NTOK, 512 / 64);
        }

        // --- depthwise conv + SiLU ---
        conv_silu_kernel<<<blocks(NTOK * D_INNER, EBLK), EBLK, 0, stream>>>(
            xz, conv_w + (size_t)i * D_INNER * D_CONV, conv_b + (size_t)i * D_INNER,
            xc, xc16);

        // --- x_proj: (8192,256) x (64pad,256)^T -> f32 xdbl (8192,64) ---
        {
            const int strips = (NTOK / 32) * 1;               // 256
            wmma_gemm_kernel<D_INNER, D_INNER, D_INNER, XP_NPAD, false>
                <<<blocks(strips, 4), 128, 0, stream>>>(
                xc16, w_xp + (size_t)i * XP_NPAD * D_INNER, xdbl, nullptr, NTOK, 1);
        }

        // --- delta = softplus(dt @ dtw^T + dtb) ---
        delta_kernel<<<blocks(NTOK * D_INNER, EBLK), EBLK, 0, stream>>>(
            xdbl, dt_w + (size_t)i * D_INNER * DT_RANK, dt_b + (size_t)i * D_INNER, dlt);

        // --- chunked selective scan (3 passes) ---
        scan_local_kernel<<<scanBlocks, EBLK, 0, stream>>>(
            xc, xdbl, dlt, A_log + (size_t)i * D_INNER * D_STATE, carA, carS);
        scan_carry_kernel<<<blocks(BATCH * D_INNER * D_STATE, EBLK), EBLK, 0, stream>>>(
            carA, carS, inSt);
        scan_final_kernel<<<scanBlocks, EBLK, 0, stream>>>(
            xc, xdbl, dlt, xz,
            A_log + (size_t)i * D_INNER * D_STATE, D_skip + (size_t)i * D_INNER,
            inSt, y16);

        // --- out_proj: (8192,256) x (128,256)^T -> f32 hb + f16 h16 ---
        {
            const int strips = (NTOK / 32) * (DIMC / 64);     // 512
            wmma_gemm_kernel<D_INNER, D_INNER, D_INNER, DIMC, true>
                <<<blocks(strips, 4), 128, 0, stream>>>(
                y16, w_out + (size_t)i * DIMC * D_INNER, hb, h16, NTOK, DIMC / 64);
        }
    }

    // final RMS norm + layout back to (B,C,H,W)
    rmsnorm_out_kernel<<<NTOK, DIMC, 0, stream>>>(hb, rms_w, out);
}
